// BertAttention_79310866088226
// MI455X (gfx1250) — matrix operand order repaired and checked by reading, not compiled, not run on hardware
//
#include <hip/hip_runtime.h>
#include <hip/hip_bf16.h>

typedef __attribute__((ext_vector_type(16))) _Float16 v16h;
typedef __attribute__((ext_vector_type(8)))  _Float16 v8h;
typedef __attribute__((ext_vector_type(4)))  _Float16 v4h;
typedef __attribute__((ext_vector_type(8)))  float    v8f;
typedef __attribute__((ext_vector_type(4)))  float    v4f;

#define HN 16      // attention heads
#define DD 1024    // model dim
#define DH 64      // head dim

static __device__ __forceinline__ v16h cat8(v8h lo, v8h hi) {
  return __builtin_shufflevector(lo, hi, 0,1,2,3,4,5,6,7,8,9,10,11,12,13,14,15);
}

static __device__ __forceinline__ v8f wmma_f16(v16h a, v16h b, v8f c) {
  // D = A(16x32 f16) x B(32x16 f16) + C(16x16 f32)
  return __builtin_amdgcn_wmma_f32_16x16x32_f16(false, a, false, b, (short)0, c,
                                                false, false);
}

// ---------------------------------------------------------------------------
// fp32 -> f16 conversion (grid-stride, 4-wide)
// ---------------------------------------------------------------------------
__global__ void cvt_f32_f16_kernel(const float* __restrict__ in,
                                   _Float16* __restrict__ out, int n4) {
  int i = blockIdx.x * blockDim.x + threadIdx.x;
  int stride = gridDim.x * blockDim.x;
  for (int idx = i; idx < n4; idx += stride) {
    v4f x = ((const v4f*)in)[idx];
    v4h h;
    h[0] = (_Float16)x[0]; h[1] = (_Float16)x[1];
    h[2] = (_Float16)x[2]; h[3] = (_Float16)x[3];
    ((v4h*)out)[idx] = h;
  }
}

// ---------------------------------------------------------------------------
// WMMA GEMM: C(16x64 per wave) = A(T x K) @ W^T (Dout x K), + bias.
// A-matrix lane layout (f16 16x32): lane holds row (lane&15), K = kb..kb+15
// with kb = (lane<16 ? 0 : 16).
// B-matrix lane layout (f16 32x16): lane holds column (lane&15) = W row, with
// K sub-range offset (lane<16 ? 0 : 16). Both are contiguous-K loads.
// mode 0: store f16 per-head (n, s, dh)       (for Q, K)
// mode 1: store f16 per-head transposed (n, dh, s)   (for V)
// mode 2: store f32 row-major (t, DD)          (for output projection)
// ---------------------------------------------------------------------------
__global__ __launch_bounds__(32) void gemm16_kernel(
    const _Float16* __restrict__ A, const _Float16* __restrict__ W,
    const float* __restrict__ bias, void* __restrict__ outp,
    int K, int Bv, int S, int mode) {
  const int lane = threadIdx.x;
  const int cidx = lane & 15;
  const int hi   = lane >> 4;
  const int t0   = blockIdx.x * 16;
  const int n0   = blockIdx.y * 64;

  const _Float16* arow = A + (size_t)(t0 + cidx) * K + hi * 16;
  const _Float16* brow = W + (size_t)(n0 + cidx) * K + hi * 16;

  v8f acc[4] = {{}, {}, {}, {}};
#pragma unroll 4
  for (int k = 0; k < K; k += 32) {
    v16h a = *(const v16h*)(arow + k);
#pragma unroll
    for (int nt = 0; nt < 4; ++nt) {
      v16h b = *(const v16h*)(brow + (size_t)nt * 16 * K + k);
      acc[nt] = wmma_f16(a, b, acc[nt]);
    }
  }

  const int rbase = hi * 8;
  if (mode == 2) {
    float* out = (float*)outp;
#pragma unroll
    for (int nt = 0; nt < 4; ++nt) {
      const int col = n0 + nt * 16 + cidx;
      const float bc = bias[col];
#pragma unroll
      for (int r = 0; r < 8; ++r)
        out[(size_t)(t0 + rbase + r) * DD + col] = acc[nt][r] + bc;
    }
  } else {
    _Float16* out = (_Float16*)outp;
#pragma unroll
    for (int nt = 0; nt < 4; ++nt) {
      const int col = n0 + nt * 16 + cidx;
      const float bc = bias[col];
      const int h = col >> 6;      // head
      const int d = col & (DH - 1);
#pragma unroll
      for (int r = 0; r < 8; ++r) {
        int t = t0 + rbase + r;
        int s = t / Bv, b = t % Bv;
        int n = b * HN + h;
        _Float16 v = (_Float16)(acc[nt][r] + bc);
        if (mode == 0)
          out[((size_t)n * S + s) * DH + d] = v;       // (n, s, dh)
        else
          out[((size_t)n * DH + d) * S + s] = v;       // (n, dh, s)
      }
    }
  }
}

// ---------------------------------------------------------------------------
// Flash attention: one wave per (head-batch n, 16-query tile).
// Streams 32 keys per iteration: 4 WMMAs for scores, online softmax with
// cross-lane row reductions, P re-layout through LDS (C-format -> A-format),
// 4 WMMAs for P @ V. No score matrix materialized (no 1/sqrt(dh) scaling,
// matching the reference).
// ---------------------------------------------------------------------------
__global__ __launch_bounds__(32) void flash_attn_kernel(
    const _Float16* __restrict__ qp,   // (N, S, DH)
    const _Float16* __restrict__ kp,   // (N, S, DH)
    const _Float16* __restrict__ vt,   // (N, DH, S)
    _Float16* __restrict__ ctxh,       // (T, DD) f16
    int S, int Bv) {
  __shared__ __align__(16) float Pld[16 * 32];

  const int n    = blockIdx.x;
  const int qt   = blockIdx.y;
  const int lane = threadIdx.x;
  const int cidx = lane & 15;
  const int hi   = lane >> 4;
  const int rbase = hi * 8;
  const int kb    = hi * 16;

  // Q tile in A-layout, K=64 as two 32-chunks
  const _Float16* qrow = qp + ((size_t)n * S + qt * 16 + cidx) * DH;
  v16h qa0 = *(const v16h*)(qrow + kb);
  v16h qa1 = *(const v16h*)(qrow + 32 + kb);

  const _Float16* kbaseP = kp + (size_t)n * S * DH;
  const _Float16* vbaseP = vt + (size_t)n * DH * (size_t)S;

  float m[8], l[8];
  v8f c0 = {}, c1 = {}, c2 = {}, c3 = {};
#pragma unroll
  for (int r = 0; r < 8; ++r) { m[r] = -1e30f; l[r] = 0.f; }

  for (int j = 0; j < S; j += 32) {
    // scores S(16 x 32) = q(16x64) @ K_tile^T
    v8f s0 = {}, s1 = {};
    {
      const _Float16* kc = kbaseP + (size_t)(j + cidx) * DH + hi * 16;
      s0 = wmma_f16(qa0, *(const v16h*)(kc),      s0);
      s0 = wmma_f16(qa1, *(const v16h*)(kc + 32), s0);
      const _Float16* kc2 = kbaseP + (size_t)(j + 16 + cidx) * DH + hi * 16;
      s1 = wmma_f16(qa0, *(const v16h*)(kc2),      s1);
      s1 = wmma_f16(qa1, *(const v16h*)(kc2 + 32), s1);
    }

    // online softmax over the 32 new columns (row = rbase + r for this lane)
    float alpha[8];
#pragma unroll
    for (int r = 0; r < 8; ++r) {
      float mx = fmaxf(s0[r], s1[r]);
      mx = fmaxf(mx, __shfl_xor(mx, 1, 32));
      mx = fmaxf(mx, __shfl_xor(mx, 2, 32));
      mx = fmaxf(mx, __shfl_xor(mx, 4, 32));
      mx = fmaxf(mx, __shfl_xor(mx, 8, 32));
      float mn = fmaxf(m[r], mx);
      alpha[r] = __expf(m[r] - mn);
      m[r] = mn;
      float p0 = __expf(s0[r] - mn);
      float p1 = __expf(s1[r] - mn);
      s0[r] = p0; s1[r] = p1;
      float rs = p0 + p1;
      rs += __shfl_xor(rs, 1, 32);
      rs += __shfl_xor(rs, 2, 32);
      rs += __shfl_xor(rs, 4, 32);
      rs += __shfl_xor(rs, 8, 32);
      l[r] = l[r] * alpha[r] + rs;
    }
#pragma unroll
    for (int r = 0; r < 8; ++r) {
      c0[r] *= alpha[r]; c1[r] *= alpha[r];
      c2[r] *= alpha[r]; c3[r] *= alpha[r];
    }

    // re-layout P: C-format (col per lane) -> A-format (row per lane) via LDS
    asm volatile("" ::: "memory");
#pragma unroll
    for (int r = 0; r < 8; ++r) {
      Pld[(rbase + r) * 32 + cidx]      = s0[r];
      Pld[(rbase + r) * 32 + 16 + cidx] = s1[r];
    }
    asm volatile("s_wait_dscnt 0" ::: "memory");  // wave DS ops are in-order
    v16h pa;
    {
      const v4f* prow = (const v4f*)(Pld + cidx * 32 + kb);
      v4f pA = prow[0], pB = prow[1], pC = prow[2], pD = prow[3];
#pragma unroll
      for (int i = 0; i < 4; ++i) {
        pa[i]      = (_Float16)pA[i];
        pa[4 + i]  = (_Float16)pB[i];
        pa[8 + i]  = (_Float16)pC[i];
        pa[12 + i] = (_Float16)pD[i];
      }
    }
    asm volatile("" ::: "memory");

    // ctx(16x64) += P(16x32) @ V(32x64); V^T layout makes B columns contiguous
    const size_t vo = (size_t)cidx * S + j + hi * 16;
    c0 = wmma_f16(pa, *(const v16h*)(vbaseP + vo),                          c0);
    c1 = wmma_f16(pa, *(const v16h*)(vbaseP + (size_t)16 * S + vo),        c1);
    c2 = wmma_f16(pa, *(const v16h*)(vbaseP + (size_t)32 * S + vo),        c2);
    c3 = wmma_f16(pa, *(const v16h*)(vbaseP + (size_t)48 * S + vo),        c3);
  }

  // normalize and store ctx as f16 (t, DD) for the output projection GEMM
  const int b = n / HN, h = n % HN;
#pragma unroll
  for (int r = 0; r < 8; ++r) {
    float inv = 1.0f / l[r];
    int srow = qt * 16 + rbase + r;
    size_t base = ((size_t)srow * Bv + b) * DD + (size_t)h * DH + cidx;
    ctxh[base + 0]  = (_Float16)(c0[r] * inv);
    ctxh[base + 16] = (_Float16)(c1[r] * inv);
    ctxh[base + 32] = (_Float16)(c2[r] * inv);
    ctxh[base + 48] = (_Float16)(c3[r] * inv);
  }
}

// ---------------------------------------------------------------------------
// residual add + LayerNorm, one block per token row
// ---------------------------------------------------------------------------
__global__ __launch_bounds__(256) void ln_resid_kernel(
    const float* __restrict__ y, const float* __restrict__ resid,
    const float* __restrict__ w, const float* __restrict__ bln,
    float* __restrict__ out) {
  const int t = blockIdx.x;
  const int tid = threadIdx.x;
  __shared__ float red[256];

  float x[4];
  float s = 0.f;
#pragma unroll
  for (int i = 0; i < 4; ++i) {
    int idx = tid + i * 256;
    x[i] = y[(size_t)t * DD + idx] + resid[(size_t)t * DD + idx];
    s += x[i];
  }
  red[tid] = s; __syncthreads();
  for (int off = 128; off > 0; off >>= 1) {
    if (tid < off) red[tid] += red[tid + off];
    __syncthreads();
  }
  float u = red[0] * (1.0f / DD);
  __syncthreads();

  float s2 = 0.f;
#pragma unroll
  for (int i = 0; i < 4; ++i) { float d = x[i] - u; s2 += d * d; }
  red[tid] = s2; __syncthreads();
  for (int off = 128; off > 0; off >>= 1) {
    if (tid < off) red[tid] += red[tid + off];
    __syncthreads();
  }
  float inv = rsqrtf(red[0] * (1.0f / DD) + 1e-12f);

#pragma unroll
  for (int i = 0; i < 4; ++i) {
    int idx = tid + i * 256;
    out[(size_t)t * DD + idx] = w[idx] * ((x[i] - u) * inv) + bln[idx];
  }
}

// ---------------------------------------------------------------------------
extern "C" void kernel_launch(void* const* d_in, const int* in_sizes, int n_in,
                              void* d_out, int out_size, void* d_ws, size_t ws_size,
                              hipStream_t stream) {
  const int S = 2048, Bv = 2, T = S * Bv, N = Bv * HN;
  const size_t TD = (size_t)T * DD;       // 4,194,304
  const size_t WW = (size_t)DD * DD;      // 1,048,576

  const float* q_in = (const float*)d_in[0];
  const float* k_in = (const float*)d_in[1];
  const float* v_in = (const float*)d_in[2];
  const float* Wq = (const float*)d_in[3];
  const float* bq = (const float*)d_in[4];
  const float* Wk = (const float*)d_in[5];
  const float* bk = (const float*)d_in[6];
  const float* Wv = (const float*)d_in[7];
  const float* bv = (const float*)d_in[8];
  const float* Wo = (const float*)d_in[9];
  const float* bo = (const float*)d_in[10];
  const float* lnw = (const float*)d_in[11];
  const float* lnb = (const float*)d_in[12];

  char* ws = (char*)d_ws;
  size_t off = 0;
  auto alloc = [&](size_t bytes) -> void* {
    void* p = ws + off;
    off += (bytes + 255) & ~(size_t)255;
    return p;
  };
  _Float16* xq_h = (_Float16*)alloc(TD * 2);
  _Float16* xk_h = (_Float16*)alloc(TD * 2);
  _Float16* xv_h = (_Float16*)alloc(TD * 2);
  _Float16* wq_h = (_Float16*)alloc(WW * 2);
  _Float16* wk_h = (_Float16*)alloc(WW * 2);
  _Float16* wv_h = (_Float16*)alloc(WW * 2);
  _Float16* wo_h = (_Float16*)alloc(WW * 2);
  _Float16* qp   = (_Float16*)alloc(TD * 2);   // (N, S, DH)
  _Float16* kpb  = (_Float16*)alloc(TD * 2);   // (N, S, DH)
  _Float16* vtp  = (_Float16*)alloc(TD * 2);   // (N, DH, S)
  _Float16* ctxh = (_Float16*)alloc(TD * 2);   // (T, DD)
  float*    ybuf = (float*)   alloc(TD * 4);   // (T, DD) f32

  // 1) fp32 -> f16 conversions
  cvt_f32_f16_kernel<<<1024, 256, 0, stream>>>(q_in, xq_h, (int)(TD / 4));
  cvt_f32_f16_kernel<<<1024, 256, 0, stream>>>(k_in, xk_h, (int)(TD / 4));
  cvt_f32_f16_kernel<<<1024, 256, 0, stream>>>(v_in, xv_h, (int)(TD / 4));
  cvt_f32_f16_kernel<<<512, 256, 0, stream>>>(Wq, wq_h, (int)(WW / 4));
  cvt_f32_f16_kernel<<<512, 256, 0, stream>>>(Wk, wk_h, (int)(WW / 4));
  cvt_f32_f16_kernel<<<512, 256, 0, stream>>>(Wv, wv_h, (int)(WW / 4));
  cvt_f32_f16_kernel<<<512, 256, 0, stream>>>(Wo, wo_h, (int)(WW / 4));

  // 2) Q/K/V projections (WMMA)
  dim3 gg(T / 16, DD / 64);
  gemm16_kernel<<<gg, 32, 0, stream>>>(xq_h, wq_h, bq, qp,  DD, Bv, S, 0);
  gemm16_kernel<<<gg, 32, 0, stream>>>(xk_h, wk_h, bk, kpb, DD, Bv, S, 0);
  gemm16_kernel<<<gg, 32, 0, stream>>>(xv_h, wv_h, bv, vtp, DD, Bv, S, 1);

  // 3) attention (WMMA flash, no score materialization)
  flash_attn_kernel<<<dim3(N, S / 16), 32, 0, stream>>>(qp, kpb, vtp, ctxh, S, Bv);

  // 4) output projection (WMMA, f32 out)
  gemm16_kernel<<<gg, 32, 0, stream>>>(ctxh, wo_h, bo, ybuf, DD, Bv, S, 2);

  // 5) residual + LayerNorm
  ln_resid_kernel<<<T, 256, 0, stream>>>(ybuf, q_in, lnw, lnb, (float*)d_out);
}
